// FormulaNet_48730698940500
// MI455X (gfx1250) — matrix operand order repaired and checked by reading, not compiled, not run on hardware
//
#include <hip/hip_runtime.h>

// ---------------------------------------------------------------------------
// FormulaNet on MI455X (gfx1250): fully fused tree-GEMM per workgroup.
//   - wave32, v_wmma_f32_16x16x32_f16 (f16 A/B, f32 accumulate)
//   - 8 batch elements per block, whole 64-leaf tree resident in LDS (250 KB)
//   - per-wave N-tile fixed -> weight (B) fragments cached in VGPRs per level
//   - scalar tile control flow (readfirstlane): EXEC stays all-ones at WMMA
//   - dual accumulator chains + double-buffered A loads: wmma waits on
//     dscnt<=4 instead of full drains, overlapping LDS latency with math
// ---------------------------------------------------------------------------

typedef __attribute__((ext_vector_type(16))) _Float16     v16h;
typedef __attribute__((ext_vector_type(8)))  float        v8f;
typedef __attribute__((ext_vector_type(4))) unsigned int  v4u;

#define TB        8                 // batch elements per block
#define NTHREADS  256               // 8 wave32
#define TSTRIDE   160               // tree A row stride (halves): 50|50|50|10pad
#define LSTRIDE   320               // leaf A row stride (halves): 6*50|20pad

// LDS layout (halves):
//  ABUF  40960  (leaf stage 128x320 / odd-level A 128x160)
//  BBUF  40960  (even-level A 256x160)
//  WPT   20480  (W_pred  N-major f16, 64x320)
//  WBT   10240  (W_bin   N-major f16, 64x160)
//  EP/EV/EO 3x5000 (embedding tables f16, 100x50)
//  BIASP/BIASB 64+64 f32
#define SMEM_BYTES (255280 + 512)

union AFrag { v4u q[2]; v16h h; };

// ---------------------------------------------------------------------------
// prep next-level A buffer: zero pad cols 150..159 of valid rows, zero the
// rows that round M up to a 16-multiple, scatter op embeddings to cols 50..99.
// All writes are disjoint from the concurrent GEMM epilogue (cols 0..49,
// 100..149 of valid rows), so no extra barrier is needed.
// ---------------------------------------------------------------------------
__device__ __forceinline__ void prep_level(_Float16* buf, int n_out, const int* op_ids,
                                           int off, int gb0, const _Float16* EO, int tid)
{
    unsigned int* bu = (unsigned int*)buf;
    const unsigned int* EOu = (const unsigned int*)EO;
    const int rows = n_out * TB;
    const int R16  = (rows + 15) & ~15;
    for (int t = tid; t < rows * 5; t += NTHREADS) {          // pad cols (u32 75..79)
        int r = t / 5, k = t % 5;
        bu[r * 80 + 75 + k] = 0u;
    }
    for (int t = tid; t < (R16 - rows) * 80; t += NTHREADS) { // round-up rows
        int r = rows + t / 80, k = t % 80;
        bu[r * 80 + k] = 0u;
    }
    for (int t = tid; t < rows * 25; t += NTHREADS) {         // op embeddings -> cols 50..99
        int r = t / 25, s = t % 25;
        int j = r >> 3, b = r & 7;
        int id = op_ids[(gb0 + b) * 63 + off + j];
        bu[r * 80 + 25 + s] = EOu[id * 25 + s];
    }
}

// Epilogue for one 16x16 D tile: bias + ReLU + f32->f16, scattered into the
// next level's concat slot.  Row m -> node j=jbase+(m>>3), batch b=m&7;
// column n<50 -> Dt[((j>>1)*8+b)*160 + (j&1)*100 + n].
__device__ __forceinline__ void store_tile(const v8f& c, int mt, int Mvalid, int jbase,
                                           int ng, float bb, int hf, _Float16* Dt)
{
    if (ng < 50) {
#pragma unroll
        for (int v = 0; v < 8; ++v) {
            const int m = mt * 16 + v + 8 * hf;   // C/D layout: M = v + 8*half
            if (m < Mvalid) {
                const int j = jbase + (m >> 3), bl = m & 7;
                float val = c[v] + bb;
                val = val > 0.f ? val : 0.f;
                Dt[((j >> 1) * TB + bl) * TSTRIDE + (j & 1) * 100 + ng] = (_Float16)val;
            }
        }
    }
}

// ---------------------------------------------------------------------------
// Tiled WMMA GEMM from LDS:  D = relu(A @ Wt^T + bias).
// A: [Mtiles*16 x KSTEPS*32] f16 (stride Astride); Wt: N-major [64 x KSTEPS*32].
// Wave w owns N-tile (w&3) and M-tiles (w>>2), (w>>2)+2, ...; pairs of M-tiles
// run as two independent accumulator chains sharing the cached B fragments,
// with a 2-deep A-fragment pipeline (loads for k+1 issued before wmma k).
// ---------------------------------------------------------------------------
template<int KSTEPS>
__device__ __forceinline__ void gemm_level(const _Float16* As, int Astride,
                                           const _Float16* Wt,
                                           const float* bias, _Float16* Dt,
                                           int Mtiles, int Mvalid, int jbase, int tid)
{
    const int wave = __builtin_amdgcn_readfirstlane(tid >> 5);  // SGPR wave id
    const int ln   = tid & 31;
    const int hf   = ln >> 4;      // lane half: 0 or 1
    const int lm   = ln & 15;      // row (A) / column (B) within tile
    const int nt   = wave & 3;     // fixed N-tile for this wave
    const int mt0  = wave >> 2;

    // Preload all B fragments for this wave's N-tile: reused by every M-tile.
    // B fragment: lane holds col nt*16+lm, K = ks*32 + 8*hf + j (+16 for j>=8)
    v16h bf[KSTEPS];
    {
        const _Float16* brow = Wt + (nt * 16 + lm) * (KSTEPS * 32) + 8 * hf;
#pragma unroll
        for (int ks = 0; ks < KSTEPS; ++ks) {
            AFrag t;
            t.q[0] = *(const v4u*)(brow + ks * 32);
            t.q[1] = *(const v4u*)(brow + ks * 32 + 16);
            bf[ks] = t.h;
        }
    }
    const int   ng = nt * 16 + lm;              // global output column
    const float bb = (ng < 50) ? bias[ng] : 0.f;

    int mt = mt0;
    // ---- pair loop: two chains, 2-deep pipelined A loads, no predication ----
    for (; mt + 2 < Mtiles; mt += 4) {
        const int mt2 = mt + 2;
        v8f c0 = {}, c1 = {};
        // A fragment: lane holds row mt*16+lm, K = ks*32 + 8*hf + j (+16 for j>=8)
        const _Float16* arow0 = As + (mt  * 16 + lm) * Astride + 8 * hf;
        const _Float16* arow1 = As + (mt2 * 16 + lm) * Astride + 8 * hf;
        AFrag a0[2], a1[2];
        a0[0].q[0] = *(const v4u*)(arow0);
        a0[0].q[1] = *(const v4u*)(arow0 + 16);
        a1[0].q[0] = *(const v4u*)(arow1);
        a1[0].q[1] = *(const v4u*)(arow1 + 16);
#pragma unroll
        for (int ks = 0; ks < KSTEPS; ++ks) {
            const int cb = ks & 1, nb = cb ^ 1;
            if (ks + 1 < KSTEPS) {          // prefetch k+1 before computing k
                a0[nb].q[0] = *(const v4u*)(arow0 + (ks + 1) * 32);
                a0[nb].q[1] = *(const v4u*)(arow0 + (ks + 1) * 32 + 16);
                a1[nb].q[0] = *(const v4u*)(arow1 + (ks + 1) * 32);
                a1[nb].q[1] = *(const v4u*)(arow1 + (ks + 1) * 32 + 16);
            }
            c0 = __builtin_amdgcn_wmma_f32_16x16x32_f16(
                     false, a0[cb].h, false, bf[ks], (short)0, c0, false, false);
            c1 = __builtin_amdgcn_wmma_f32_16x16x32_f16(
                     false, a1[cb].h, false, bf[ks], (short)0, c1, false, false);
        }
        store_tile(c0, mt,  Mvalid, jbase, ng, bb, hf, Dt);
        store_tile(c1, mt2, Mvalid, jbase, ng, bb, hf, Dt);
    }
    // ---- scalar tail: at most one remaining M-tile for this wave ----
    if (mt < Mtiles) {
        v8f c0 = {};
        const _Float16* arow0 = As + (mt * 16 + lm) * Astride + 8 * hf;
        AFrag a0[2];
        a0[0].q[0] = *(const v4u*)(arow0);
        a0[0].q[1] = *(const v4u*)(arow0 + 16);
#pragma unroll
        for (int ks = 0; ks < KSTEPS; ++ks) {
            const int cb = ks & 1, nb = cb ^ 1;
            if (ks + 1 < KSTEPS) {
                a0[nb].q[0] = *(const v4u*)(arow0 + (ks + 1) * 32);
                a0[nb].q[1] = *(const v4u*)(arow0 + (ks + 1) * 32 + 16);
            }
            c0 = __builtin_amdgcn_wmma_f32_16x16x32_f16(
                     false, a0[cb].h, false, bf[ks], (short)0, c0, false, false);
        }
        store_tile(c0, mt, Mvalid, jbase, ng, bb, hf, Dt);
    }
}

__global__ __launch_bounds__(NTHREADS)
void formula_net_kernel(const int* __restrict__ pred_ids,  // (B,64)
                        const int* __restrict__ arg_ids,   // (B,64,5)
                        const int* __restrict__ op_ids,    // (B,63)
                        const float* __restrict__ emb_pred,
                        const float* __restrict__ emb_var,
                        const float* __restrict__ emb_op,
                        const float* __restrict__ W_pred,  // (50,300)
                        const float* __restrict__ b_pred,
                        const float* __restrict__ W_bin,   // (50,150)
                        const float* __restrict__ b_bin,
                        const float* __restrict__ W_final, // (155,50)
                        const float* __restrict__ b_final,
                        float* __restrict__ out)           // (B,155)
{
    extern __shared__ char smem[];
    _Float16* ABUF = (_Float16*)smem;          // 40960 halves
    _Float16* BBUF = ABUF + 40960;             // 40960
    _Float16* WPT  = BBUF + 40960;             // 20480  (64 x 320)
    _Float16* WBT  = WPT + 20480;              // 10240  (64 x 160)
    _Float16* EP   = WBT + 10240;              // 5000
    _Float16* EV   = EP + 5000;                // 5000
    _Float16* EO   = EV + 5000;                // 5000
    float*    BIASP = (float*)(EO + 5000);     // 64
    float*    BIASB = BIASP + 64;              // 64

    const int tid = threadIdx.x;
    const int gb0 = blockIdx.x * TB;

    unsigned int* WPTu = (unsigned int*)WPT;
    unsigned int* WBTu = (unsigned int*)WBT;

    // ---- Phase A: zero weight pads, convert tables, biases (disjoint) ----
    for (int t = tid; t < 10240 + 5120; t += NTHREADS) {
        if (t < 10240) WPTu[t] = 0u; else WBTu[t - 10240] = 0u;
    }
    for (int t = tid; t < 15000; t += NTHREADS) {
        int which = t / 5000, e = t % 5000;
        const float* s = (which == 0) ? emb_pred : (which == 1) ? emb_var : emb_op;
        _Float16*    d = (which == 0) ? EP       : (which == 1) ? EV      : EO;
        d[e] = (_Float16)s[e];
    }
    for (int t = tid; t < 128; t += NTHREADS) {
        if (t < 64) BIASP[t] = (t < 50) ? b_pred[t] : 0.f;
        else { int k = t - 64; BIASB[k] = (k < 50) ? b_bin[k] : 0.f; }
    }
    __syncthreads();

    // ---- Phase B: fill f16 N-major weights; prep level-0 A buffer ----
    for (int t = tid; t < 7500; t += NTHREADS) {          // W_pred -> WPT[64][320]
        int n = t / 150, s = t % 150;
        union { unsigned int u; _Float16 h[2]; } pk;
        pk.h[0] = (_Float16)W_pred[n * 300 + 2 * s];
        pk.h[1] = (_Float16)W_pred[n * 300 + 2 * s + 1];
        WPTu[n * 160 + s] = pk.u;
    }
    for (int t = tid; t < 3750; t += NTHREADS) {          // W_bin -> WBT[64][160]
        int n = t / 75, s = t % 75;
        union { unsigned int u; _Float16 h[2]; } pk;
        pk.h[0] = (_Float16)W_bin[n * 150 + 2 * s];
        pk.h[1] = (_Float16)W_bin[n * 150 + 2 * s + 1];
        WBTu[n * 80 + s] = pk.u;
    }
    prep_level(BBUF, 32, op_ids, 0, gb0, EO, tid);        // A(0): 256 rows
    __syncthreads();

    // ---- Leaf level: 64 nodes in 4 chunks of 16 (M=128, K=320, N=64) ----
    unsigned int* ABUFu = (unsigned int*)ABUF;
    const unsigned int* EPu = (const unsigned int*)EP;
    const unsigned int* EVu = (const unsigned int*)EV;
    for (int c = 0; c < 4; ++c) {
        // stage A: row r = (node-c*16)*8 + b, cols = [pred|arg0..arg4|pad20]
        for (int t = tid; t < 128 * 6; t += NTHREADS) {
            int r = t / 6, seg = t % 6;
            int q = r >> 3, bl = r & 7;
            int node = c * 16 + q, gb = gb0 + bl;
            const unsigned int* src;
            if (seg == 0) src = EPu + pred_ids[gb * 64 + node] * 25;
            else          src = EVu + arg_ids[(gb * 64 + node) * 5 + seg - 1] * 25;
            unsigned int* dst = ABUFu + r * 160 + seg * 25;
#pragma unroll
            for (int s = 0; s < 25; ++s) dst[s] = src[s];
        }
        for (int t = tid; t < 128 * 10; t += NTHREADS) {  // pad cols 300..319
            int r = t / 10, k = t % 10;
            ABUFu[r * 160 + 150 + k] = 0u;
        }
        __syncthreads();
        gemm_level<10>(ABUF, LSTRIDE, WPT, BIASP, BBUF,
                       /*Mtiles*/8, /*Mvalid*/128, /*jbase*/c * 16, tid);
        __syncthreads();
    }

    // ---- Tree levels: 64->32->16->8->4->2->1 (M=n_out*8, K=160, N=64) ----
    const int OFF[6] = {0, 32, 48, 56, 60, 62};
    for (int i = 0; i < 6; ++i) {
        _Float16* cur = (i & 1) ? ABUF : BBUF;
        _Float16* nxt = (i & 1) ? BBUF : ABUF;
        const int n_out = 64 >> (i + 1);
        if (i < 5) prep_level(nxt, 64 >> (i + 2), op_ids, OFF[i + 1], gb0, EO, tid);
        const int M = n_out * TB;
        gemm_level<5>(cur, TSTRIDE, WBT, BIASB, nxt,
                      (M + 15) / 16, M, 0, tid);
        __syncthreads();
    }

    // ---- Final linear: x (8x50, in BBUF rows 0..7) @ W_final^T + b_final ----
    const _Float16* X = BBUF;
    for (int t = tid; t < TB * 155; t += NTHREADS) {
        int bl = t / 155, o = t % 155;
        float acc = b_final[o];
        const float* wr = W_final + o * 50;
#pragma unroll 10
        for (int k = 0; k < 50; ++k) acc += (float)X[bl * TSTRIDE + k] * wr[k];
        out[(gb0 + bl) * 155 + o] = acc;
    }
}

extern "C" void kernel_launch(void* const* d_in, const int* in_sizes, int n_in,
                              void* d_out, int out_size, void* d_ws, size_t ws_size,
                              hipStream_t stream) {
    (void)in_sizes; (void)n_in; (void)out_size; (void)d_ws; (void)ws_size;
    const int*   pred_ids = (const int*)d_in[0];
    const int*   arg_ids  = (const int*)d_in[1];
    const int*   op_ids   = (const int*)d_in[2];
    const float* emb_pred = (const float*)d_in[3];
    const float* emb_var  = (const float*)d_in[4];
    const float* emb_op   = (const float*)d_in[5];
    const float* W_pred   = (const float*)d_in[6];
    const float* b_pred   = (const float*)d_in[7];
    const float* W_bin    = (const float*)d_in[8];
    const float* b_bin    = (const float*)d_in[9];
    const float* W_final  = (const float*)d_in[10];
    const float* b_final  = (const float*)d_in[11];
    float* out = (float*)d_out;

    (void)hipFuncSetAttribute((const void*)formula_net_kernel,
                              hipFuncAttributeMaxDynamicSharedMemorySize, SMEM_BYTES);
    formula_net_kernel<<<4096 / TB, NTHREADS, SMEM_BYTES, stream>>>(
        pred_ids, arg_ids, op_ids, emb_pred, emb_var, emb_op,
        W_pred, b_pred, W_bin, b_bin, W_final, b_final, out);
}
